// GreedyInference_73383811220062
// MI455X (gfx1250) — matrix operand order repaired and checked by reading, not compiled, not run on hardware
//
#include <hip/hip_runtime.h>
#include <hip/hip_bf16.h>
#include <math.h>

// Problem constants (match reference)
#define NB 32        // batch
#define TT 1000      // time steps
#define DD 640       // encoder dim
#define HH 640       // hidden
#define EE 640       // embed dim
#define JJ 640       // joint dim
#define VV 1024      // vocab (BLANK index)
#define VO 1025      // vocab + blank
#define G4 2560      // 4*H
#define KC 1280      // concat(emb,h) K dim
#define SYMS 5
#define BLANKI 1024
#define NWG 20       // decode workgroups (hidden slice = 640/20 = 32)
#define BLOCK 256
#define JSL 32       // hidden columns per WG

typedef __bf16 bf16_t;
typedef __attribute__((ext_vector_type(16))) __bf16 v16bf;
typedef __attribute__((ext_vector_type(8)))  float  v8f;
typedef __attribute__((ext_vector_type(4)))  unsigned int v4u;

union Frag32B { v4u q[2]; v16bf v; };

static __device__ __forceinline__ unsigned short f2bfu(float f) {
  bf16_t h = (bf16_t)f;
  return __builtin_bit_cast(unsigned short, h);
}

// B-fragment (32x16 bf16): lane = column; lanes 0-15 K=k..k+15, lanes 16-31 K=k+16..k+31.
// Caller bakes the 16*(lane>>4) K-offset into the row pointer; 16 contiguous bf16 here.
static __device__ __forceinline__ v16bf load_frag_contig(const unsigned short* p) {
  Frag32B f;
  f.q[0] = *(const v4u*)(p);
  f.q[1] = *(const v4u*)(p + 8);
  return f.v;
}

// A-fragment (16x32 bf16): lane L%16 = row M; L/16 selects K-half:
// elems 0..15 = K k0+16*kh..k0+16*kh+15 (contiguous)
static __device__ __forceinline__ v16bf load_fragA(const unsigned short* row, int k0, int kh) {
  Frag32B f;
  f.q[0] = *(const v4u*)(row + k0 + 16 * kh);
  f.q[1] = *(const v4u*)(row + k0 + 16 * kh + 8);
  return f.v;
}

static __device__ __forceinline__ v8f wmma_bf16(v16bf a, v16bf b, v8f c) {
  return __builtin_amdgcn_wmma_f32_16x16x32_bf16(false, a, false, b, (short)0, c, false, false);
}

static __device__ __forceinline__ float sigm(float x) { return 1.0f / (1.0f + expf(-x)); }

// Monotone float->u32 key (NaN-free data): larger float => larger key
static __device__ __forceinline__ unsigned sortable_f32(float f) {
  unsigned u = __builtin_bit_cast(unsigned, f);
  return (u & 0x80000000u) ? ~u : (u | 0x80000000u);
}

static __device__ __forceinline__ unsigned long long shflxor64(unsigned long long v, int m) {
  unsigned lo = (unsigned)v;
  unsigned hi = (unsigned)(v >> 32);
  lo = __shfl_xor(lo, m, 32);
  hi = __shfl_xor(hi, m, 32);
  return (((unsigned long long)hi) << 32) | (unsigned long long)lo;
}

// Sense-reversal grid barrier across NWG resident workgroups.
static __device__ __forceinline__ void grid_sync(unsigned* sync) {
  __syncthreads();
  if (threadIdx.x == 0) {
    __threadfence();
    unsigned g = __hip_atomic_load(&sync[1], __ATOMIC_ACQUIRE, __HIP_MEMORY_SCOPE_AGENT);
    unsigned a = __hip_atomic_fetch_add(&sync[0], 1u, __ATOMIC_ACQ_REL, __HIP_MEMORY_SCOPE_AGENT);
    if (a == (unsigned)(NWG - 1)) {
      __hip_atomic_store(&sync[0], 0u, __ATOMIC_RELAXED, __HIP_MEMORY_SCOPE_AGENT);
      __hip_atomic_fetch_add(&sync[1], 1u, __ATOMIC_ACQ_REL, __HIP_MEMORY_SCOPE_AGENT);
    } else {
      while (__hip_atomic_load(&sync[1], __ATOMIC_ACQUIRE, __HIP_MEMORY_SCOPE_AGENT) == g) {
        __builtin_amdgcn_s_sleep(1);
      }
    }
    __threadfence();
  }
  __syncthreads();
}

// ---------------- one-time prep kernels ----------------

// enc [B][D][T] f32 -> enc_bf [B*T][D] bf16 (row-major A matrix)
__global__ __launch_bounds__(BLOCK) void k_enc_transpose(const float* __restrict__ enc,
                                                         unsigned short* __restrict__ enc_bf) {
  int i = blockIdx.x * BLOCK + threadIdx.x;
  if (i >= NB * TT * DD) return;
  int d = i % DD;
  int bt = i / DD;
  int t = bt % TT;
  int b = bt / TT;
  enc_bf[i] = f2bfu(enc[((long long)b * DD + d) * TT + t]);
}

// wcat_t[n][k] = (k<E ? Wx[k][n] : Wh[k-E][n]) as bf16 ; n in [0,2560), k in [0,1280)
__global__ __launch_bounds__(BLOCK) void k_build_wcat(const float* __restrict__ Wx,
                                                      const float* __restrict__ Wh,
                                                      unsigned short* __restrict__ wcat) {
  int i = blockIdx.x * BLOCK + threadIdx.x;
  if (i >= G4 * KC) return;
  int n = i / KC;
  int k = i % KC;
  float v = (k < EE) ? Wx[(long long)k * G4 + n] : Wh[(long long)(k - EE) * G4 + n];
  wcat[i] = f2bfu(v);
}

// Wt[n][k] = W[k][n] as bf16 (W is [K][N] row-major)
__global__ __launch_bounds__(BLOCK) void k_transpose_w(const float* __restrict__ W,
                                                       unsigned short* __restrict__ Wt,
                                                       int K, int N) {
  int i = blockIdx.x * BLOCK + threadIdx.x;
  if (i >= N * K) return;
  int n = i / K;
  int k = i % K;
  Wt[i] = f2bfu(W[(long long)k * N + n]);
}

// enc_proj[b*T+t][j]: one 16x32 (2 N-tiles) block per wave, K split in two chains
// => 4 independent WMMA accumulator chains per wave, A fragment shared by both N-tiles.
__global__ __launch_bounds__(256) void k_enc_proj(const unsigned short* __restrict__ enc_bf,
                                                  const unsigned short* __restrict__ wenc_t,
                                                  float* __restrict__ enc_proj) {
  int wave = threadIdx.x >> 5;
  int lane = threadIdx.x & 31;
  int l16 = lane & 15;
  int kh = lane >> 4;
  int tile = blockIdx.x * 8 + wave;           // 2000 M-tiles * 20 N-pairs = 40000
  if (tile >= 2000 * 20) return;
  int np = tile % 20;
  int mt = tile / 20;
  int m0 = mt * 16, n0 = np * 32;
  const unsigned short* arow = enc_bf + (long long)(m0 + l16) * DD;
  const unsigned short* b0 = wenc_t + (long long)(n0 + l16) * DD + 16 * kh;
  const unsigned short* b1 = wenc_t + (long long)(n0 + 16 + l16) * DD + 16 * kh;
  v8f acc00 = {}, acc01 = {}, acc10 = {}, acc11 = {};
  for (int k = 0; k < DD; k += 64) {
    v16bf a0 = load_fragA(arow, k, kh);
    v16bf a1 = load_fragA(arow, k + 32, kh);
    acc00 = wmma_bf16(a0, load_frag_contig(b0 + k), acc00);
    acc01 = wmma_bf16(a1, load_frag_contig(b0 + k + 32), acc01);
    acc10 = wmma_bf16(a0, load_frag_contig(b1 + k), acc10);
    acc11 = wmma_bf16(a1, load_frag_contig(b1 + k + 32), acc11);
  }
  v8f accA = acc00 + acc01;
  v8f accB = acc10 + acc11;
  int n = n0 + l16;
#pragma unroll
  for (int r = 0; r < 8; ++r) {
    int m = m0 + kh * 8 + r;
    enc_proj[(long long)m * JJ + n] = accA[r];
    enc_proj[(long long)m * JJ + n + 16] = accB[r];
  }
}

__global__ __launch_bounds__(BLOCK) void k_init(float* h, float* c, int* lastb, int* alive,
                                                int* emitf, unsigned long long* amax,
                                                unsigned* sync, float* out) {
  int base = blockIdx.x * BLOCK + threadIdx.x;
  int stride = gridDim.x * BLOCK;
  for (int i = base; i < NB * HH; i += stride) { h[i] = 0.0f; c[i] = 0.0f; }
  for (int i = base; i < NB; i += stride) {
    lastb[i] = BLANKI; alive[i] = 0; emitf[i] = 0; amax[i] = 0ull;
  }
  if (base == 0) { sync[0] = 0u; sync[1] = 0u; sync[2] = 0u; }
  for (int i = base; i < NB * TT * SYMS; i += stride) out[i] = (float)BLANKI;  // blank prefill
}

// ---------------- persistent decode kernel ----------------

__global__ __launch_bounds__(BLOCK) void k_decode(
    const float* __restrict__ embed, const float* __restrict__ b_lstm,
    const float* __restrict__ b_joint, const float* __restrict__ b_out,
    const int* __restrict__ enc_lens,
    const unsigned short* __restrict__ wcat_t,   // [2560][1280]
    const unsigned short* __restrict__ wpred_t,  // [640][640]
    const unsigned short* __restrict__ wout_t,   // [1025][640]
    const float* __restrict__ enc_proj,          // [B*T][J]
    unsigned short* act_bf,                      // [32][1280]  concat(emb,h) bf16
    unsigned short* hnew_bf,                     // [32][640]
    unsigned short* z_bf,                        // [32][640]
    float* h_state, float* c_state, float* hnew_f, float* cnew_f,
    unsigned long long* amax, int* lastb, int* alive, int* emitf,
    unsigned* sync, float* out) {
  const int wg = blockIdx.x;
  const int tid = threadIdx.x;
  const int wave = tid >> 5;
  const int lane = tid & 31;
  const int l16 = lane & 15;
  const int kh = lane >> 4;
  const int j0 = wg * JSL;   // this WG's hidden/joint column slice [j0, j0+32)

  __shared__ float gates_lds[NB][4 * JSL];                    // 16 KB
  __shared__ __align__(16) unsigned short z_lds[NB * JJ];     // 40 KB, async-staged
  __shared__ int s_flag;

  for (int t = 0; t < TT; ++t) {
    for (int s = 0; s < SYMS; ++s) {
      // ---- P1: commit pending state for our slice + build bf16 activation matrix ----
      for (int idx = tid; idx < NB * JSL; idx += BLOCK) {
        int b = idx >> 5, jj = idx & 31;
        int j = j0 + jj;
        float hv = h_state[b * HH + j];
        if (emitf[b]) {
          hv = hnew_f[b * HH + j];
          h_state[b * HH + j] = hv;
          c_state[b * HH + j] = cnew_f[b * HH + j];
        }
        act_bf[b * KC + EE + j] = f2bfu(hv);
        int l = lastb[b];
        float ev = (l == BLANKI) ? 0.0f : embed[l * EE + j];
        act_bf[b * KC + j] = f2bfu(ev);
      }
      grid_sync(sync);

      // ---- P2: gates = concat(emb,h) @ Wcat ; wave = (mt, gate); both 16-col subtiles
      // share the A fragment; K split in two => 4 independent WMMA chains per wave ----
      {
        int mt = wave & 1;
        int g = wave >> 1;
        int m0 = mt * 16;
        int nb0 = g * HH + j0;
        const unsigned short* arow = act_bf + (m0 + l16) * KC;
        const unsigned short* b0 = wcat_t + (nb0 + l16) * KC + 16 * kh;
        const unsigned short* b1 = wcat_t + (nb0 + 16 + l16) * KC + 16 * kh;
        v8f acc00 = {}, acc01 = {}, acc10 = {}, acc11 = {};
        for (int k = 0; k < KC; k += 64) {
          v16bf a0 = load_fragA(arow, k, kh);
          v16bf a1 = load_fragA(arow, k + 32, kh);
          acc00 = wmma_bf16(a0, load_frag_contig(b0 + k), acc00);
          acc01 = wmma_bf16(a1, load_frag_contig(b0 + k + 32), acc01);
          acc10 = wmma_bf16(a0, load_frag_contig(b1 + k), acc10);
          acc11 = wmma_bf16(a1, load_frag_contig(b1 + k + 32), acc11);
        }
        v8f accA = acc00 + acc01;   // columns [j0, j0+16)
        v8f accB = acc10 + acc11;   // columns [j0+16, j0+32)
        int ncol = g * JSL + l16;
#pragma unroll
        for (int r = 0; r < 8; ++r) {
          gates_lds[m0 + kh * 8 + r][ncol] = accA[r];
          gates_lds[m0 + kh * 8 + r][ncol + 16] = accB[r];
        }
      }
      __syncthreads();
      for (int idx = tid; idx < NB * JSL; idx += BLOCK) {
        int b = idx >> 5, jj = idx & 31;
        int j = j0 + jj;
        float gi = gates_lds[b][0 * JSL + jj] + b_lstm[0 * HH + j];
        float gf = gates_lds[b][1 * JSL + jj] + b_lstm[1 * HH + j];
        float gg = gates_lds[b][2 * JSL + jj] + b_lstm[2 * HH + j];
        float go = gates_lds[b][3 * JSL + jj] + b_lstm[3 * HH + j];
        float cn = sigm(gf) * c_state[b * HH + j] + sigm(gi) * tanhf(gg);
        float hn = sigm(go) * tanhf(cn);
        cnew_f[b * HH + j] = cn;
        hnew_f[b * HH + j] = hn;
        hnew_bf[b * HH + j] = f2bfu(hn);
      }
      grid_sync(sync);

      // ---- P3: z = tanh(enc_proj_t + h_new @ Wpred + b_joint) ; K-split dual chains ----
      if (wave < 4) {
        int mt = wave & 1;
        int nt = wave >> 1;
        int m0 = mt * 16;
        int n0 = j0 + nt * 16;
        const unsigned short* arow = hnew_bf + (m0 + l16) * HH;
        const unsigned short* brow = wpred_t + (n0 + l16) * HH + 16 * kh;
        v8f acc0 = {}, acc1 = {};
        for (int k = 0; k < HH; k += 64) {
          acc0 = wmma_bf16(load_fragA(arow, k, kh), load_frag_contig(brow + k), acc0);
          acc1 = wmma_bf16(load_fragA(arow, k + 32, kh), load_frag_contig(brow + k + 32), acc1);
        }
        v8f acc = acc0 + acc1;
        int n = n0 + l16;
#pragma unroll
        for (int r = 0; r < 8; ++r) {
          int m = m0 + kh * 8 + r;  // batch row
          float v = acc[r] + enc_proj[(m * TT + t) * JJ + n] + b_joint[n];
          z_bf[m * JJ + n] = f2bfu(tanhf(v));
        }
      }
      grid_sync(sync);

      // ---- P4: stage z into LDS via async copy, then logits = z @ Wout + b_out ----
      {
        // 40 KB coalesced async copy: 16B per lane per op, tracked by ASYNCcnt.
        for (int ch = tid; ch < (NB * JJ) / 8; ch += BLOCK) {
          unsigned lds_addr = (unsigned)(unsigned long long)(&z_lds[ch * 8]);
          unsigned long long gaddr = (unsigned long long)(const unsigned short*)(z_bf + ch * 8);
          asm volatile("global_load_async_to_lds_b128 %0, %1, off"
                       :: "v"(lds_addr), "v"(gaddr) : "memory");
        }
        asm volatile("s_wait_asynccnt 0" ::: "memory");
        __syncthreads();

        int gw = wg * 8 + wave;                      // 160 waves, 130 tile jobs
        for (int jid = gw; jid < 2 * 65; jid += NWG * 8) {
          int mt = jid & 1;
          int nt = jid >> 1;                         // 0..64 (65*16 covers 1025 cols)
          int m0 = mt * 16;
          int n = nt * 16 + l16;
          int nc = (n > VV) ? VV : n;                // clamp for loads
          const unsigned short* arow = &z_lds[(m0 + l16) * JJ];
          const unsigned short* brow = wout_t + nc * JJ + 16 * kh;
          v8f acc0 = {}, acc1 = {};
          for (int k = 0; k < JJ; k += 64) {
            acc0 = wmma_bf16(load_fragA(arow, k, kh), load_frag_contig(brow + k), acc0);
            acc1 = wmma_bf16(load_fragA(arow, k + 32, kh), load_frag_contig(brow + k + 32), acc1);
          }
          v8f acc = acc0 + acc1;
          unsigned long long key[8];
          bool valid = (n <= VV);
#pragma unroll
          for (int r = 0; r < 8; ++r) {
            float v = acc[r] + b_out[nc];
            // key: larger logit wins; ties -> smaller index (matches jnp.argmax)
            key[r] = valid ? ((((unsigned long long)sortable_f32(v)) << 32) |
                              (unsigned long long)(0xFFFFFFFFu - (unsigned)n))
                           : 0ull;
          }
#pragma unroll
          for (int msk = 1; msk < 16; msk <<= 1) {
#pragma unroll
            for (int r = 0; r < 8; ++r) {
              unsigned long long o = shflxor64(key[r], msk);
              key[r] = (o > key[r]) ? o : key[r];
            }
          }
          if (l16 == 0) {
#pragma unroll
            for (int r = 0; r < 8; ++r)
              atomicMax(&amax[m0 + kh * 8 + r], key[r]);
          }
        }
      }
      grid_sync(sync);

      // ---- P5: WG0 decides emit / writes labels / updates last,alive ----
      if (wg == 0 && tid < NB) {
        int b = tid;
        unsigned long long key = amax[b];
        int sym = (int)(0xFFFFFFFFu - (unsigned)(key & 0xFFFFFFFFull));
        int aliveCur = (s == 0) ? ((t < enc_lens[b]) ? 1 : 0) : alive[b];
        int em = (aliveCur && sym != BLANKI) ? 1 : 0;
        out[b * (TT * SYMS) + t * SYMS + s] = (float)(em ? sym : BLANKI);
        if (em) lastb[b] = sym;
        alive[b] = em;
        emitf[b] = em;
        amax[b] = 0ull;
        unsigned long long bal = __ballot(em);
        if (b == 0)
          __hip_atomic_store(&sync[2], (bal == 0ull) ? 1u : 0u,
                             __ATOMIC_RELAXED, __HIP_MEMORY_SCOPE_AGENT);
      }
      grid_sync(sync);

      if (tid == 0)
        s_flag = (int)__hip_atomic_load(&sync[2], __ATOMIC_RELAXED, __HIP_MEMORY_SCOPE_AGENT);
      __syncthreads();
      if (s_flag) break;   // nobody emitted: rest of this t's slots stay BLANK, state frozen
    }
  }

  // final commit of pending state for our slice + write cache_rnn_state [2][B][H]
  for (int idx = tid; idx < NB * JSL; idx += BLOCK) {
    int b = idx >> 5, jj = idx & 31;
    int j = j0 + jj;
    float hv = h_state[b * HH + j];
    float cv = c_state[b * HH + j];
    if (emitf[b]) { hv = hnew_f[b * HH + j]; cv = cnew_f[b * HH + j]; }
    out[NB * TT * SYMS + b * HH + j] = hv;
    out[NB * TT * SYMS + NB * HH + b * HH + j] = cv;
  }
}

// ---------------- host launcher ----------------

extern "C" void kernel_launch(void* const* d_in, const int* in_sizes, int n_in,
                              void* d_out, int out_size, void* d_ws, size_t ws_size,
                              hipStream_t stream) {
  (void)in_sizes; (void)n_in; (void)out_size; (void)ws_size;
  const float* enc      = (const float*)d_in[0];
  const int*   enc_lens = (const int*)d_in[1];
  const float* embed    = (const float*)d_in[2];
  const float* Wx       = (const float*)d_in[3];
  const float* Wh       = (const float*)d_in[4];
  const float* b_lstm   = (const float*)d_in[5];
  const float* Wenc     = (const float*)d_in[6];
  const float* Wpred    = (const float*)d_in[7];
  const float* b_joint  = (const float*)d_in[8];
  const float* Wout     = (const float*)d_in[9];
  const float* b_out    = (const float*)d_in[10];
  float* out = (float*)d_out;

  char* ws = (char*)d_ws;
  size_t off = 0;
  auto carve = [&](size_t bytes) -> char* {
    char* p = ws + off;
    off += (bytes + 255) & ~(size_t)255;
    return p;
  };
  unsigned short* enc_bf   = (unsigned short*)carve((size_t)NB * TT * DD * 2);
  float*          enc_proj = (float*)carve((size_t)NB * TT * JJ * 4);
  unsigned short* wcat_t   = (unsigned short*)carve((size_t)G4 * KC * 2);
  unsigned short* wpred_t  = (unsigned short*)carve((size_t)JJ * HH * 2);
  unsigned short* wout_t   = (unsigned short*)carve((size_t)VO * JJ * 2);
  unsigned short* wenc_t   = (unsigned short*)carve((size_t)JJ * DD * 2);
  unsigned short* act_bf   = (unsigned short*)carve((size_t)NB * KC * 2);
  unsigned short* hnew_bf  = (unsigned short*)carve((size_t)NB * HH * 2);
  unsigned short* z_bf     = (unsigned short*)carve((size_t)NB * JJ * 2);
  float* h_state = (float*)carve((size_t)NB * HH * 4);
  float* c_state = (float*)carve((size_t)NB * HH * 4);
  float* hnew_f  = (float*)carve((size_t)NB * HH * 4);
  float* cnew_f  = (float*)carve((size_t)NB * HH * 4);
  unsigned long long* amax = (unsigned long long*)carve((size_t)NB * 8);
  int* lastb = (int*)carve((size_t)NB * 4);
  int* alive = (int*)carve((size_t)NB * 4);
  int* emitf = (int*)carve((size_t)NB * 4);
  unsigned* sync = (unsigned*)carve(64);

  k_enc_transpose<<<(NB * TT * DD + BLOCK - 1) / BLOCK, BLOCK, 0, stream>>>(enc, enc_bf);
  k_build_wcat<<<(G4 * KC + BLOCK - 1) / BLOCK, BLOCK, 0, stream>>>(Wx, Wh, wcat_t);
  k_transpose_w<<<(JJ * DD + BLOCK - 1) / BLOCK, BLOCK, 0, stream>>>(Wenc, wenc_t, DD, JJ);
  k_transpose_w<<<(JJ * HH + BLOCK - 1) / BLOCK, BLOCK, 0, stream>>>(Wpred, wpred_t, HH, JJ);
  k_transpose_w<<<(VO * JJ + BLOCK - 1) / BLOCK, BLOCK, 0, stream>>>(Wout, wout_t, JJ, VO);
  k_enc_proj<<<5000, 256, 0, stream>>>(enc_bf, wenc_t, enc_proj);
  k_init<<<1024, BLOCK, 0, stream>>>(h_state, c_state, lastb, alive, emitf, amax, sync, out);
  k_decode<<<NWG, BLOCK, 0, stream>>>(embed, b_lstm, b_joint, b_out, enc_lens,
                                      wcat_t, wpred_t, wout_t, enc_proj,
                                      act_bf, hnew_bf, z_bf,
                                      h_state, c_state, hnew_f, cnew_f,
                                      amax, lastb, alive, emitf, sync, out);
}
